// TaxiCollectiveCriticWithCostAndBiasVPN_33328946217220
// MI455X (gfx1250) — compile-verified, matrix operand order checked by reading, not run on hardware
//
#include <hip/hip_runtime.h>
#include <math.h>

typedef __attribute__((ext_vector_type(16))) _Float16 v16h;
typedef __attribute__((ext_vector_type(8)))  float    v8f;

// Problem constants
#define BZ_   8192
#define Z_    81
#define H_    48
#define ZZ_   6561          // Z*Z
#define OBSW_ 210           // H + 2Z
#define NSCAL 8000.0f

// Output layout (flat f32 concat, element offsets)
static constexpr size_t O_SYM = 0;                            // (B,)
static constexpr size_t O_VN  = 8192;                         // (B,Z,Z)
static constexpr size_t O_CF  = O_VN  + (size_t)BZ_ * ZZ_;    // (B,Z)
static constexpr size_t O_IMM = O_CF  + (size_t)BZ_ * Z_;     // (B,Z)
static constexpr size_t O_D   = O_IMM + (size_t)BZ_ * Z_;     // (B,Z)
static constexpr size_t O_FTW = O_D   + (size_t)BZ_ * Z_;     // (B,Z)
static constexpr size_t O_FV  = O_FTW + (size_t)BZ_ * Z_;     // (B,Z)  (temp: served)
static constexpr size_t O_NX  = O_FV  + (size_t)BZ_ * Z_;     // (B,Z)  (temp: x)

// ---------------------------------------------------------------------------
// Kernel 0 (optional): one-shot WV f32 -> f16 conversion into workspace.
// Halves the hot loop's B-fetch bytes and removes all inner-loop cvts.
// ---------------------------------------------------------------------------
__global__ __launch_bounds__(256) void wv_to_f16_kernel(
    const float* __restrict__ WV, _Float16* __restrict__ WVh)
{
  int i = blockIdx.x * 256 + threadIdx.x;
  if (i < H_ * ZZ_) WVh[i] = (_Float16)WV[i];
}

// ---------------------------------------------------------------------------
// Kernel 1: one pass over action_count (215 MB, the big read) + small GEMVs.
// Produces immediate, d, ftw (final) and stashes served -> out[O_FV],
// x -> out[O_NX] for kernel 2 to consume and overwrite.
// ---------------------------------------------------------------------------
__global__ __launch_bounds__(96) void prep_kernel(
    const float* __restrict__ obs, const float* __restrict__ ac,
    const float* __restrict__ cost, const float* __restrict__ tw,
    const float* __restrict__ Wd, const float* __restrict__ Wf,
    float* __restrict__ out)
{
  __shared__ float tp[H_];
  const int b = blockIdx.x;
  const int t = threadIdx.x;
  if (t < H_) tp[t] = obs[(size_t)b * OBSW_ + t] * NSCAL;
  __syncthreads();

  if (t < Z_) {
    // x[k] = N * sum_z ac[b,z,k]; cost_sum[k] = -N * sum_z ac[b,z,k]*cost[z,k]
    const float* acb = ac + (size_t)b * ZZ_ + t;
    float sx = 0.f, sc = 0.f;
    #pragma unroll 3
    for (int z = 0; z < Z_; ++z) {
      float a = acb[z * Z_];
      sx += a;
      sc = fmaf(a, cost[z * Z_ + t], sc);
    }
    float x        = sx * NSCAL;
    float cost_sum = -NSCAL * sc;
    float dem      = obs[(size_t)b * OBSW_ + (H_ + Z_) + t] * NSCAL;
    float served   = fminf(x, dem);
    float imm      = fmaf(tw[t], served, cost_sum);

    // d = relu(tp @ Wd), ftw = relu(tp @ Wf)
    float dd = 0.f, ff = 0.f;
    #pragma unroll
    for (int h = 0; h < H_; ++h) {
      float th = tp[h];
      dd = fmaf(th, Wd[h * Z_ + t], dd);
      ff = fmaf(th, Wf[h * Z_ + t], ff);
    }
    dd = fmaxf(dd, 0.f);
    ff = fmaxf(ff, 0.f);

    size_t gi = (size_t)b * Z_ + t;
    out[O_IMM + gi] = imm;
    out[O_D   + gi] = dd;
    out[O_FTW + gi] = ff;
    out[O_FV  + gi] = served;  // temp stash, overwritten by kernel 2
    out[O_NX  + gi] = x;       // temp stash, overwritten by kernel 2
  }
}

// ---------------------------------------------------------------------------
// Kernel 2: WMMA GEMM (16 batch rows x 6561 cols, K=48 in f16) fused with
// softmax(axis=2), V_norm write (the big 215 MB store), customer_flow einsum,
// and all tail outputs. 192 threads = 6 wave32; wave w owns n-tile w of the
// 96-wide (>=81) column slab for each softmax group z.
// BT = _Float16 when WV was pre-converted into workspace, else float.
// ---------------------------------------------------------------------------
template <typename BT>
__global__ __launch_bounds__(192) void fused_kernel(
    const float* __restrict__ obs, const BT* __restrict__ WV,
    const float* __restrict__ bV,  float* __restrict__ out)
{
  __shared__ _Float16 aLds[16][64];     // A tile, f16, K padded 48->64
  __shared__ float    vbuf[16][100];    // V slab (16 x 96 used), padded stride
  __shared__ float    sLds[16][84];     // served tile
  __shared__ float    cfb [16][84];     // customer_flow accumulator
  __shared__ float    red [16][12];     // row reductions
  __shared__ float    rowmax[16];
  __shared__ float    rowinv[16];

  const int tid  = threadIdx.x;
  const int wave = tid >> 5;            // 0..5
  const int lane = tid & 31;
  const int nlo  = lane & 15;
  const int hi   = (lane >> 4) & 1;
  const int b0   = blockIdx.x * 16;

  // ---- stage A (f16) + served, zero cf ----
  for (int i = tid; i < 16 * 64; i += 192) {
    int m = i >> 6, k = i & 63;
    float v = (k < H_) ? obs[(size_t)(b0 + m) * OBSW_ + k] * NSCAL : 0.f;
    aLds[m][k] = (_Float16)v;
  }
  for (int i = tid; i < 16 * Z_; i += 192) {
    int m = i / Z_, k = i - m * Z_;
    sLds[m][k] = out[O_FV + (size_t)(b0 + m) * Z_ + k];
  }
  for (int i = tid; i < 16 * 84; i += 192) cfb[i / 84][i - (i / 84) * 84] = 0.f;
  __syncthreads();

  // ---- build per-wave A fragments (doc 7.12.2: 16-bit A 16x32 layout) ----
  // half t of lane L: K = t + (t>=8 ? 8 : 0) + (hi ? 8 : 0)
  v16h afrag0, afrag1;
  {
    const int mA = nlo;
    #pragma unroll
    for (int t = 0; t < 16; ++t) {
      int k = t + ((t >= 8) ? 8 : 0) + (hi ? 8 : 0);
      afrag0[t] = aLds[mA][k];
      afrag1[t] = aLds[mA][k + 32];     // K 32..47 real, 48..63 zero-padded
    }
  }

  const int col = wave * 16 + nlo;      // 0..95 within slab

  for (int z = 0; z < Z_; ++z) {
    // ---- load B fragments for this wave's 16 columns (clamped OOB) ----
    int j = z * Z_ + col; if (j > ZZ_ - 1) j = ZZ_ - 1;
    const BT* wvp = WV + j;
    v16h bfrag0, bfrag1;
    #pragma unroll
    for (int t = 0; t < 16; ++t) {
      int k = t + ((t >= 8) ? 8 : 0) + (hi ? 8 : 0);
      bfrag0[t] = (_Float16)wvp[(size_t)k * ZZ_];
      int k1 = k + 32;
      bfrag1[t] = (k1 < H_) ? (_Float16)wvp[(size_t)k1 * ZZ_] : (_Float16)0.f;
    }
    if (z + 1 < Z_) {                   // warm next slab (WV is L2-resident)
      int jn = (z + 1) * Z_ + col; if (jn > ZZ_ - 1) jn = ZZ_ - 1;
      __builtin_prefetch(WV + jn, 0, 1);
    }

    // ---- 16x16 tile: two K=32 WMMA steps (K=48 zero-padded) ----
    v8f c = {};
    c = __builtin_amdgcn_wmma_f32_16x16x32_f16(false, afrag0, false, bfrag0,
                                               (short)0, c, false, false);
    c = __builtin_amdgcn_wmma_f32_16x16x32_f16(false, afrag1, false, bfrag1,
                                               (short)0, c, false, false);

    // ---- deposit per f32 C/D layout: elem (M = v + 8*hi, N = nlo) ----
    float bvv = bV[j];
    #pragma unroll
    for (int v = 0; v < 8; ++v) vbuf[v + 8 * hi][col] = c[v] + bvv;
    __syncthreads();

    // ---- softmax over cols 0..80 of each of the 16 rows ----
    const int m = tid / 12, slot = tid - m * 12;
    float pm = -INFINITY;
    for (int k = slot; k < Z_; k += 12) pm = fmaxf(pm, vbuf[m][k]);
    red[m][slot] = pm;
    __syncthreads();
    if (tid < 16) {
      float r = red[tid][0];
      #pragma unroll
      for (int s = 1; s < 12; ++s) r = fmaxf(r, red[tid][s]);
      rowmax[tid] = r;
    }
    __syncthreads();
    float rm = rowmax[m];
    float ps = 0.f;
    for (int k = slot; k < Z_; k += 12) {
      float e = __expf(vbuf[m][k] - rm);
      vbuf[m][k] = e;
      ps += e;
    }
    red[m][slot] = ps;
    __syncthreads();
    if (tid < 16) {
      float s = red[tid][0];
      #pragma unroll
      for (int q = 1; q < 12; ++q) s += red[tid][q];
      rowinv[tid] = 1.0f / s;
    }
    __syncthreads();

    // ---- coalesced V_norm store + customer_flow accumulation ----
    for (int i = tid; i < 16 * Z_; i += 192) {
      int mm = i / Z_, k = i - mm * Z_;
      float vn = vbuf[mm][k] * rowinv[mm];
      out[O_VN + (size_t)(b0 + mm) * ZZ_ + z * Z_ + k] = vn;
      cfb[mm][k] = fmaf(sLds[mm][z], vn, cfb[mm][k]);
    }
    __syncthreads();
  }

  // ---- tail: customer_flow, next_x, future_vals, symbolic_val ----
  for (int i = tid; i < 16 * Z_; i += 192) {
    int m = i / Z_, k = i - m * Z_;
    size_t gi = (size_t)(b0 + m) * Z_ + k;
    float served = sLds[m][k];
    float x      = out[O_NX + gi];          // stashed by kernel 1
    float cflow  = cfb[m][k];
    float nx     = x - served + cflow;
    float fv     = fminf(nx * out[O_FTW + gi], out[O_D + gi]);
    out[O_CF + gi] = cflow;
    out[O_NX + gi] = nx;
    out[O_FV + gi] = fv;
    cfb[m][k] = out[O_IMM + gi] + fv;       // reuse LDS for final reduction
  }
  __syncthreads();
  {
    const int m = tid / 12, slot = tid - m * 12;
    float ps = 0.f;
    for (int k = slot; k < Z_; k += 12) ps += cfb[m][k];
    red[m][slot] = ps;
    __syncthreads();
    if (tid < 16) {
      float s = red[tid][0];
      #pragma unroll
      for (int q = 1; q < 12; ++q) s += red[tid][q];
      out[O_SYM + b0 + tid] = s;
    }
  }
}

// ---------------------------------------------------------------------------
extern "C" void kernel_launch(void* const* d_in, const int* in_sizes, int n_in,
                              void* d_out, int out_size, void* d_ws, size_t ws_size,
                              hipStream_t stream) {
  (void)in_sizes; (void)n_in; (void)out_size;
  const float* obs  = (const float*)d_in[0];
  const float* ac   = (const float*)d_in[1];
  const float* cost = (const float*)d_in[2];
  const float* tw   = (const float*)d_in[3];
  const float* WV   = (const float*)d_in[4];
  const float* bV   = (const float*)d_in[5];
  const float* Wd   = (const float*)d_in[6];
  const float* Wf   = (const float*)d_in[7];
  float* out = (float*)d_out;

  prep_kernel<<<BZ_, 96, 0, stream>>>(obs, ac, cost, tw, Wd, Wf, out);

  const size_t wvh_bytes = (size_t)H_ * ZZ_ * sizeof(_Float16);
  if (ws_size >= wvh_bytes) {
    _Float16* WVh = (_Float16*)d_ws;
    wv_to_f16_kernel<<<(H_ * ZZ_ + 255) / 256, 256, 0, stream>>>(WV, WVh);
    fused_kernel<_Float16><<<BZ_ / 16, 192, 0, stream>>>(obs, WVh, bV, out);
  } else {
    fused_kernel<float><<<BZ_ / 16, 192, 0, stream>>>(obs, WV, bV, out);
  }
}
